// LSTM_63591285785268
// MI455X (gfx1250) — compile-verified, matrix operand order checked by reading, not run on hardware
//
#include <hip/hip_runtime.h>
#include <hip/hip_bf16.h>

#define Hdim 1024
#define Bdim 64
#define Sdim 512

// Tiling: each block covers 32 output columns (2 WMMA N-tiles), 8 waves =
// (4 M-tiles) x (2 N-halves)  ->  2 waves per SIMD32 for latency hiding.
#define NCOLS    32                  // columns per block
#define KC       128                 // K elements per staged chunk
#define NCHUNK   (Hdim / KC)         // 8
#define PITCH_E  (KC + 8)            // row pitch in elements (136 -> 272B, avoids bank conflicts)
#define BUFELEMS (3 * NCOLS * PITCH_E)  // one buffer: 3 gates x 32 cols = 13056 elems (26112 B)

typedef __attribute__((ext_vector_type(16))) __bf16 v16bf;
typedef __attribute__((ext_vector_type(8)))  __bf16 v8bf;
typedef __attribute__((ext_vector_type(8)))  float  v8f;

// Load a 16-element bf16 fragment as two 16-byte chunks at p and p+16 elements
// (matches 16-bit WMMA A/B VGPR layout: lane holds K = {off..off+7, off+16..off+23}).
__device__ __forceinline__ v16bf load_frag16(const __bf16* p) {
    v8bf lo = *(const v8bf*)(p);
    v8bf hi = *(const v8bf*)(p + 16);
    v16bf r;
#pragma unroll
    for (int i = 0; i < 8; ++i) { r[i] = lo[i]; r[i + 8] = hi[i]; }
    return r;
}

__device__ __forceinline__ float sigmoidf_fast(float x) {
    return 1.0f / (1.0f + __expf(-x));
}

// ---- one-time: convert f32 weights -> bf16 (6 MB total; lives in the 192MB L2) ----
__global__ void rnn_cvt_weights(const float* __restrict__ W, const float* __restrict__ U,
                                const float* __restrict__ V,
                                __bf16* __restrict__ Wb, __bf16* __restrict__ Ub,
                                __bf16* __restrict__ Vb) {
    int i = blockIdx.x * blockDim.x + threadIdx.x;   // H*H elements
    Wb[i] = (__bf16)W[i];
    Ub[i] = (__bf16)U[i];
    Vb[i] = (__bf16)V[i];
}

// ---- one-time: zero recurrent state ----
__global__ void rnn_init_state(float* __restrict__ p1, float* __restrict__ p2) {
    int i = blockIdx.x * blockDim.x + threadIdx.x;   // B*H elements
    p1[i] = 0.0f;
    p2[i] = 0.0f;
}

// ---- per step, phase 1: xi = emb[:,t,:] + p2 ; stash bf16(xi) and sigmoid(xi) ----
__global__ void rnn_xi_kernel(const float* __restrict__ emb, const float* __restrict__ p2,
                              __bf16* __restrict__ xi_bf, float* __restrict__ xsig, int t) {
    int i = blockIdx.x * blockDim.x + threadIdx.x;   // B*H elements
    int b = i >> 10;            // /H
    int k = i & (Hdim - 1);
    float x = emb[(size_t)b * Sdim * Hdim + (size_t)t * Hdim + k] + p2[i];
    xi_bf[i] = (__bf16)x;
    xsig[i]  = sigmoidf_fast(x);
}

// Issue the async global->LDS copy of one K-chunk of the 3 weight tiles.
// 3 gates x 32 rows x 128 K x 2B = 24576 B = 1536 b128 transfers / 256 threads = 6 each.
__device__ __forceinline__ void stage_chunk(const __bf16* __restrict__ Wb,
                                            const __bf16* __restrict__ Ub,
                                            const __bf16* __restrict__ Vb,
                                            int blk, int chunk, int buf,
                                            unsigned smem_base, int tid) {
#pragma unroll
    for (int i = 0; i < 6; ++i) {
        int j   = tid + 256 * i;     // 0..1535
        int seg = j & 15;            // 16B segment within a row (8 bf16)
        int rg  = j >> 4;            // 0..95 : gate*32 + row
        int g   = rg >> 5;           // 0..2
        int row = rg & 31;
        const __bf16* srcm = (g == 0) ? Wb : (g == 1) ? Ub : Vb;
        const __bf16* src  = srcm + (size_t)(blk * NCOLS + row) * Hdim + chunk * KC + seg * 8;
        unsigned dst = smem_base + (unsigned)(buf * (BUFELEMS * 2) + rg * (PITCH_E * 2) + seg * 16);
        // tracked by ASYNCcnt; per-lane: LDS[dst] = MEM[src] (16 bytes)
        asm volatile("global_load_async_to_lds_b128 %0, %1, off"
                     :: "v"(dst), "v"(src) : "memory");
    }
}

// ---- per step, phase 2: three GEMMs via bf16 WMMA + fused gate math ----
// grid.x = H/32 = 32 blocks, block = 256 threads = 8 waves:
//   wave>>1 = M-tile (batch rows), wave&1 = which 16-column half of the block.
// Weight tiles staged global->LDS via async DMA, double buffered; 2 waves per
// SIMD32 interleave so s_wait_dscnt stalls of one wave hide behind the other.
__global__ void __launch_bounds__(256)
rnn_gate_kernel(const __bf16* __restrict__ Wb, const __bf16* __restrict__ Ub,
                const __bf16* __restrict__ Vb,
                const float* __restrict__ bW, const float* __restrict__ bU,
                const float* __restrict__ bV,
                const __bf16* __restrict__ xi_bf, const float* __restrict__ xsig,
                float* __restrict__ p1, float* __restrict__ p2,
                float* __restrict__ out, int t) {
    __shared__ __bf16 sB[2 * BUFELEMS];

    const int tid   = threadIdx.x;
    const int lane  = tid & 31;
    const int wave  = tid >> 5;                // 0..7
    const int mtile = wave >> 1;               // 0..3 (batch-row tile)
    const int nhalf = wave & 1;                // 0..1 (16-col half within block)
    const int blk   = blockIdx.x;              // 0..31

    const int nsub  = lane & 15;               // column within 16-wide tile
    const int khalf = lane >> 4;               // 0/1 (16-bit fragment layout half)
    const int col   = blk * NCOLS + nhalf * 16 + nsub;   // output column n
    const int arow  = mtile * 16 + nsub;       // A row (batch index striped)
    const int brow  = nhalf * 16 + nsub;       // row within the staged 32-col tile

    const __bf16* aptr = xi_bf + (size_t)arow * Hdim;
    const unsigned smem_base = (unsigned)(uintptr_t)&sB[0];   // low 32 bits = LDS offset

    v8f accF = {}; v8f accI = {}; v8f accG = {};

    // prologue: stage chunk 0 into buffer 0
    stage_chunk(Wb, Ub, Vb, blk, 0, 0, smem_base, tid);

    for (int ck = 0; ck < NCHUNK; ++ck) {
        // my async copies for chunk ck have landed
        asm volatile("s_wait_asynccnt 0x0" ::: "memory");
        // everyone's copies landed + nobody still reads the other buffer
        __syncthreads();
        if (ck + 1 < NCHUNK)
            stage_chunk(Wb, Ub, Vb, blk, ck + 1, (ck + 1) & 1, smem_base, tid);

        const __bf16* bbase = &sB[(ck & 1) * BUFELEMS + brow * PITCH_E + khalf * 8];
#pragma unroll
        for (int ks = 0; ks < KC; ks += 32) {
            v16bf a  = load_frag16(aptr + ck * KC + ks + khalf * 8);
            v16bf fw = load_frag16(bbase + 0 * NCOLS * PITCH_E + ks);   // ds_load_b128 x2
            v16bf fu = load_frag16(bbase + 1 * NCOLS * PITCH_E + ks);
            v16bf fv = load_frag16(bbase + 2 * NCOLS * PITCH_E + ks);
            accF = __builtin_amdgcn_wmma_f32_16x16x32_bf16(false, a, false, fw,
                                                           (short)0, accF, false, false);
            accI = __builtin_amdgcn_wmma_f32_16x16x32_bf16(false, a, false, fu,
                                                           (short)0, accI, false, false);
            accG = __builtin_amdgcn_wmma_f32_16x16x32_bf16(false, a, false, fv,
                                                           (short)0, accG, false, false);
        }
    }

    const float biasF = bW[col];
    const float biasI = bU[col];
    const float biasG = bV[col];

    // C/D layout: lanes 0-15 hold M = v, lanes 16-31 hold M = v+8 (per VGPR v).
#pragma unroll
    for (int v = 0; v < 8; ++v) {
        const int b   = mtile * 16 + khalf * 8 + v;   // batch row
        const int idx = b * Hdim + col;
        float f  = sigmoidf_fast(accF[v] + biasF);
        float ig = sigmoidf_fast(accI[v] + biasI);
        float g  = tanhf(accG[v] + biasG);
        float p1n = f * p1[idx] + ig * g;
        p1[idx] = p1n;
        float p2n = xsig[idx] * tanhf(p1n);
        p2[idx] = p2n;
        out[(size_t)b * Sdim * Hdim + (size_t)t * Hdim + col] = p2n;
    }
}

extern "C" void kernel_launch(void* const* d_in, const int* in_sizes, int n_in,
                              void* d_out, int out_size, void* d_ws, size_t ws_size,
                              hipStream_t stream) {
    (void)in_sizes; (void)n_in; (void)out_size; (void)ws_size;

    const float* emb = (const float*)d_in[0];
    const float* W   = (const float*)d_in[1];
    const float* bW  = (const float*)d_in[2];
    const float* U   = (const float*)d_in[3];
    const float* bU  = (const float*)d_in[4];
    const float* V   = (const float*)d_in[5];
    const float* bV  = (const float*)d_in[6];
    float* out = (float*)d_out;

    // Workspace layout
    char* ws = (char*)d_ws;
    __bf16* Wb   = (__bf16*)ws;  ws += (size_t)Hdim * Hdim * sizeof(__bf16);
    __bf16* Ub   = (__bf16*)ws;  ws += (size_t)Hdim * Hdim * sizeof(__bf16);
    __bf16* Vb   = (__bf16*)ws;  ws += (size_t)Hdim * Hdim * sizeof(__bf16);
    float*  p1   = (float*)ws;   ws += (size_t)Bdim * Hdim * sizeof(float);
    float*  p2   = (float*)ws;   ws += (size_t)Bdim * Hdim * sizeof(float);
    float*  xsig = (float*)ws;   ws += (size_t)Bdim * Hdim * sizeof(float);
    __bf16* xi   = (__bf16*)ws;  ws += (size_t)Bdim * Hdim * sizeof(__bf16);

    // One-time per call (deterministic): weight conversion + state init
    rnn_cvt_weights<<<(Hdim * Hdim) / 256, 256, 0, stream>>>(W, U, V, Wb, Ub, Vb);
    rnn_init_state<<<(Bdim * Hdim) / 256, 256, 0, stream>>>(p1, p2);

    // Serial scan over time; stream order provides the device-wide sync.
    for (int t = 0; t < Sdim; ++t) {
        rnn_xi_kernel<<<(Bdim * Hdim) / 256, 256, 0, stream>>>(emb, p2, xi, xsig, t);
        rnn_gate_kernel<<<Hdim / NCOLS, 256, 0, stream>>>(Wb, Ub, Vb, bW, bU, bV,
                                                          xi, xsig, p1, p2, out, t);
    }
}
